// MultiHeadMemory_45337674776981
// MI455X (gfx1250) — matrix operand order repaired and checked by reading, not compiled, not run on hardware
//
#include <hip/hip_runtime.h>
#include <hip/hip_bf16.h>

typedef __bf16 v16bf __attribute__((ext_vector_type(16)));
typedef __bf16 v8bf  __attribute__((ext_vector_type(8)));
typedef float  v8f   __attribute__((ext_vector_type(8)));
typedef float  v4f   __attribute__((ext_vector_type(4)));
typedef int    v4i   __attribute__((ext_vector_type(4)));

#define Hh 8
#define Mm 64
#define Dd 128
#define KDk 128
#define VDv 128
#define Nn 262144

#define AS1 __attribute__((address_space(1)))
#define AS3 __attribute__((address_space(3)))

#if defined(__has_builtin)
#if __has_builtin(__builtin_amdgcn_global_load_async_to_lds_b128) && \
    __has_builtin(__builtin_amdgcn_s_wait_asynccnt)
#define HAVE_ASYNC 1
#endif
#endif
#ifndef HAVE_ASYNC
#define HAVE_ASYNC 0
#endif

// ---------- helpers ----------------------------------------------------------

__device__ __forceinline__ unsigned pkbf(float a, float b) {
    union { __bf16 h[2]; unsigned u; } z;
    z.h[0] = (__bf16)a; z.h[1] = (__bf16)b;
    return z.u;
}

// A-fragment (16x32 bf16): lane holds A[row][c0..c0+15],
// c0 = kblock*32 + 16*half. Two 16B chunks.
__device__ __forceinline__ v16bf load_a16(const __bf16* row, int c0) {
    union { v16bf v; v8bf h2[2]; } u;
    u.h2[0] = *(const v8bf*)(row + c0);
    u.h2[1] = *(const v8bf*)(row + c0 + 8);
    return u.v;
}

// Convert two adjacent f32 D-tiles (rows [0,16) and [16,32) of a 32-row K block)
// into a bf16 B-fragment for the next WMMA. D layout: lane holds D[r+8*hh][lane&15]
// for VGPR r. B, in the k order load_a16 gives A, has lane half hh hold k = 16*hh + 0..15 packed in
// pairs. Own half supplies 8 rows; the other 8 come from lane^16.
__device__ __forceinline__ v16bf dpair_to_bfrag(v8f tLo, v8f tHi, int hh) {
    unsigned own[4], oth[4];
#pragma unroll
    for (int j = 0; j < 4; ++j) {
        float a0 = hh ? tHi[2*j]   : tLo[2*j];
        float a1 = hh ? tHi[2*j+1] : tLo[2*j+1];
        float b0 = hh ? tLo[2*j]   : tHi[2*j];
        float b1 = hh ? tLo[2*j+1] : tHi[2*j+1];
        own[j] = pkbf(a0, a1);     // pairs from this lane's own tile (tile 2b+hh)
        oth[j] = pkbf(b0, b1);     // pairs the partner half needs (tile 2b+1-hh)
    }
    union { v16bf v; unsigned u[8]; } r;
#pragma unroll
    for (int j = 0; j < 4; ++j) {
        unsigned rec = (unsigned)__shfl_xor((int)oth[j], 16, 32);
        r.u[j]     = hh ? rec    : own[j];
        r.u[4 + j] = hh ? own[j] : rec;
    }
    return r.v;
}

// B-fragment for k built directly from a global f32 row (k row == Bt row).
// k is streamed once -> non-temporal loads keep L2 free for the hot constants.
__device__ __forceinline__ v16bf k_bfrag(const float* krow, int base) {
    union { v16bf v; unsigned u[8]; } r;
#pragma unroll
    for (int q = 0; q < 4; ++q) {
        v4f f = __builtin_nontemporal_load((const v4f*)(krow + base + q * 4));
        r.u[2*q]     = pkbf(f[0], f[1]);
        r.u[2*q + 1] = pkbf(f[2], f[3]);
    }
    return r.v;
}

// Stage one head's constants (mem_key 16KB + mem_valT 16KB) into an LDS buffer.
__device__ __forceinline__ void issue_head(const __bf16* gk, const __bf16* gv,
                                           __bf16* lb, int tid) {
#if HAVE_ASYNC
#pragma unroll
    for (int i = 0; i < 4; ++i) {           // 4 async b128 per thread (mem_key)
        int j = tid + i * 256;
        __builtin_amdgcn_global_load_async_to_lds_b128(
            (AS1 v4i*)((const char*)gk + j * 16),
            (AS3 v4i*)((char*)lb + j * 16), 0, 0);
    }
#pragma unroll
    for (int i = 0; i < 4; ++i) {           // 4 async b128 per thread (mem_valT)
        int j = tid + i * 256;
        __builtin_amdgcn_global_load_async_to_lds_b128(
            (AS1 v4i*)((const char*)gv + j * 16),
            (AS3 v4i*)((char*)lb + 16384 + j * 16), 0, 0);
    }
#else
    const uint4* sk = (const uint4*)gk;
    const uint4* sv = (const uint4*)gv;
    uint4* d = (uint4*)lb;
#pragma unroll
    for (int i = 0; i < 4; ++i) {
        int j = tid + i * 256;
        d[j] = sk[j];
        d[1024 + j] = sv[j];
    }
#endif
}

// ---------- kernel 1: per-head memory keys (softmaxed) and transposed values --

__global__ __launch_bounds__(128) void mhm_pre(
    const float* __restrict__ mems, const float* __restrict__ Wk,
    const float* __restrict__ bk,   const float* __restrict__ Wv,
    const float* __restrict__ bv,   __bf16* __restrict__ wsKey,
    __bf16* __restrict__ wsValT) {
    const int hm = blockIdx.x;           // h*64 + m
    const int h  = hm >> 6;
    const int m  = hm & 63;
    const int tid = threadIdx.x;         // 0..127 -> column (kk or v)
    const int lane = tid & 31, wv = tid >> 5;

    __shared__ float md[Dd];
    __shared__ float red[8];

    md[tid] = mems[(size_t)hm * Dd + tid];
    __syncthreads();

    float ak = bk[h * KDk + tid];
    float av = bv[h * VDv + tid];
    const float* wkc = Wk + (size_t)h * Dd * KDk + tid;
    const float* wvc = Wv + (size_t)h * Dd * VDv + tid;
#pragma unroll 4
    for (int d = 0; d < Dd; ++d) {
        float x = md[d];
        ak = fmaf(x, wkc[d * KDk], ak);
        av = fmaf(x, wvc[d * VDv], av);
    }

    // softmax of ak over the 128 columns
    float mx = ak;
#pragma unroll
    for (int off = 16; off >= 1; off >>= 1) mx = fmaxf(mx, __shfl_xor(mx, off, 32));
    if (lane == 0) red[wv] = mx;
    __syncthreads();
    mx = fmaxf(fmaxf(red[0], red[1]), fmaxf(red[2], red[3]));
    float e = __expf(ak - mx);
    float s = e;
#pragma unroll
    for (int off = 16; off >= 1; off >>= 1) s += __shfl_xor(s, off, 32);
    if (lane == 0) red[4 + wv] = s;
    __syncthreads();
    s = red[4] + red[5] + red[6] + red[7];

    wsKey[(size_t)hm * KDk + tid] = (__bf16)(e / s);            // [h][m][kk]
    wsValT[((size_t)h * VDv + tid) * Mm + m] = (__bf16)av;      // [h][v][m]
}

// ---------- kernel 2: WfT[h][c][v] = Wf[h*128+v][c] in bf16 -------------------

__global__ __launch_bounds__(256) void mhm_wft(const float* __restrict__ Wf,
                                               __bf16* __restrict__ wsWfT) {
    int o = blockIdx.x * 256 + threadIdx.x;     // < 8*128*128 = 131072
    int h = o >> 14;
    int c = (o >> 7) & 127;
    int v = o & 127;
    wsWfT[o] = (__bf16)Wf[(size_t)(h * VDv + v) * VDv + c];
}

// ---------- kernel 3: fused attention + head projection -----------------------

__global__ __launch_bounds__(256) void mhm_main(
    const float* __restrict__ kq, const __bf16* __restrict__ wsKey,
    const __bf16* __restrict__ wsValT, const __bf16* __restrict__ wsWfT,
    const float* __restrict__ bf, float* __restrict__ out) {
    // double-buffered per-head constants: [buf][ mem_key 8192 | mem_valT 8192 ]
    __shared__ __align__(16) __bf16 cbuf[2 * 16384];   // 64 KB

    const int tid  = threadIdx.x;
    const int wave = tid >> 5;
    const int lane = tid & 31;
    const int hh   = (lane >> 4) & 1;     // wave half
    const int ln   = lane & 15;
    const int n0   = blockIdx.x * 128;
    const int nrow = n0 + wave * 16 + ln; // this lane's query row (B column)

    // This lane's k row feeds the GEMM1 B-fragments directly (Bt row == k row).
    const float* krow = kq + (size_t)nrow * KDk;
    v16bf bkf[4];
#pragma unroll
    for (int b = 0; b < 4; ++b) bkf[b] = k_bfrag(krow, b * 32 + hh * 16);

    v8f facc[8];
#pragma unroll
    for (int i = 0; i < 8; ++i) facc[i] = v8f{};

    // prologue: start staging head 0 into buffer 0
    issue_head(wsKey, wsValT, cbuf, tid);

    for (int h = 0; h < Hh; ++h) {
        const __bf16* akey  = cbuf + (h & 1) * 16384;
        const __bf16* avalT = akey + 8192;

        __syncthreads();   // everyone done with buffer (h+1)&1 from head h-1
        if (h + 1 < Hh)
            issue_head(wsKey + (size_t)(h + 1) * 8192,
                       wsValT + (size_t)(h + 1) * 8192,
                       cbuf + ((h + 1) & 1) * 16384, tid);
#if HAVE_ASYNC
        if (h + 1 < Hh)
            __builtin_amdgcn_s_wait_asynccnt(8);   // head h's 8 ops retired (in-order)
        else
            __builtin_amdgcn_s_wait_asynccnt(0);
#endif
        __syncthreads();   // all waves' portions of head h landed in LDS

        // ---- GEMM1: att^T[m][n] = mem_key (A) x k^T (B), K = 128 ----
        v8f p[4];
#pragma unroll
        for (int mt = 0; mt < 4; ++mt) {
            v8f acc = {};
#pragma unroll
            for (int b = 0; b < 4; ++b) {
                v16bf a = load_a16(akey + (mt * 16 + ln) * KDk, b * 32 + hh * 16);
                acc = __builtin_amdgcn_wmma_f32_16x16x32_bf16(
                    false, a, false, bkf[b], (short)0, acc, false, false);
            }
            p[mt] = acc;
        }

        // ---- softmax over m (per lane: 32 values here + 32 in lane^16) ----
        float mx = p[0][0];
#pragma unroll
        for (int t = 0; t < 4; ++t)
#pragma unroll
            for (int r = 0; r < 8; ++r) mx = fmaxf(mx, p[t][r]);
        mx = fmaxf(mx, __shfl_xor(mx, 16, 32));
        float s = 0.f;
#pragma unroll
        for (int t = 0; t < 4; ++t)
#pragma unroll
            for (int r = 0; r < 8; ++r) {
                float e = __expf(p[t][r] - mx);
                p[t][r] = e;
                s += e;
            }
        s += __shfl_xor(s, 16, 32);
        float inv = 1.0f / s;
#pragma unroll
        for (int t = 0; t < 4; ++t)
#pragma unroll
            for (int r = 0; r < 8; ++r) p[t][r] *= inv;

        // att_w D-tiles -> B-fragments (two 32-row K blocks over m)
        v16bf bp[2];
        bp[0] = dpair_to_bfrag(p[0], p[1], hh);
        bp[1] = dpair_to_bfrag(p[2], p[3], hh);

        // ---- GEMM2: out^T[v][n] = mem_val^T (A) x att_w (B), K = 64 ----
        v8f o8[8];
#pragma unroll
        for (int vt = 0; vt < 8; ++vt) {
            v8f acc = {};
#pragma unroll
            for (int b = 0; b < 2; ++b) {
                v16bf a = load_a16(avalT + (vt * 16 + ln) * Mm, b * 32 + hh * 16);
                acc = __builtin_amdgcn_wmma_f32_16x16x32_bf16(
                    false, a, false, bp[b], (short)0, acc, false, false);
            }
            o8[vt] = acc;
        }

        // out^T D-tiles -> B-fragments (four 32-row K blocks over v)
        v16bf bo[4];
#pragma unroll
        for (int b = 0; b < 4; ++b) bo[b] = dpair_to_bfrag(o8[2 * b], o8[2 * b + 1], hh);

        // ---- GEMM3: final^T[c][n] += WfT (A) x out (B), K = 128, sum heads ----
        const __bf16* wslice = wsWfT + (size_t)h * VDv * VDv;   // [c][v], L2-hot
#pragma unroll
        for (int ct = 0; ct < 8; ++ct) {
            const __bf16* arow = wslice + (ct * 16 + ln) * VDv;
#pragma unroll
            for (int b = 0; b < 4; ++b) {
                v16bf a = load_a16(arow, b * 32 + hh * 16);
                facc[ct] = __builtin_amdgcn_wmma_f32_16x16x32_bf16(
                    false, a, false, bo[b], (short)0, facc[ct], false, false);
            }
        }
    }

    // store: lane holds final^T[c = ct*16 + 8*hh + r][nrow], add bias.
    // Output is streamed once -> non-temporal stores.
    float* orow = out + (size_t)nrow * VDv;
#pragma unroll
    for (int ct = 0; ct < 8; ++ct) {
        int c = ct * 16 + hh * 8;
        float4 blo = *(const float4*)(bf + c);
        float4 bhi = *(const float4*)(bf + c + 4);
        v4f lo, hi;
        lo[0] = facc[ct][0] + blo.x; lo[1] = facc[ct][1] + blo.y;
        lo[2] = facc[ct][2] + blo.z; lo[3] = facc[ct][3] + blo.w;
        hi[0] = facc[ct][4] + bhi.x; hi[1] = facc[ct][5] + bhi.y;
        hi[2] = facc[ct][6] + bhi.z; hi[3] = facc[ct][7] + bhi.w;
        __builtin_nontemporal_store(lo, (v4f*)(orow + c));
        __builtin_nontemporal_store(hi, (v4f*)(orow + c + 4));
    }
}

// ---------- launch ------------------------------------------------------------

extern "C" void kernel_launch(void* const* d_in, const int* in_sizes, int n_in,
                              void* d_out, int out_size, void* d_ws, size_t ws_size,
                              hipStream_t stream) {
    const float* k    = (const float*)d_in[0];
    const float* mems = (const float*)d_in[1];
    const float* Wk   = (const float*)d_in[2];
    const float* bk   = (const float*)d_in[3];
    const float* Wv   = (const float*)d_in[4];
    const float* bv   = (const float*)d_in[5];
    const float* Wf   = (const float*)d_in[6];
    const float* bf   = (const float*)d_in[7];
    float* out = (float*)d_out;

    __bf16* wsKey  = (__bf16*)d_ws;                       // [H][M][KD]  65536
    __bf16* wsValT = wsKey + (size_t)Hh * Mm * KDk;       // [H][VD][M]  65536
    __bf16* wsWfT  = wsValT + (size_t)Hh * VDv * Mm;      // [H][VD][VD] 131072

    mhm_pre<<<Hh * Mm, 128, 0, stream>>>(mems, Wk, bk, Wv, bv, wsKey, wsValT);
    mhm_wft<<<(Hh * VDv * VDv) / 256, 256, 0, stream>>>(Wf, wsWfT);
    mhm_main<<<Nn / 128, 256, 0, stream>>>(k, wsKey, wsValT, wsWfT, bf, out);
}
